// DWAttentionV2_33079838113965
// MI455X (gfx1250) — compile-verified, matrix operand order checked by reading, not run on hardware
//
#include <hip/hip_runtime.h>
#include <hip/hip_bf16.h>

// ---------------------------------------------------------------------------
// MI455X (gfx1250, wave32) implementation of the LAConv windowed attention.
// Heavy math: v_wmma_f32_16x16x32_bf16 (fp32 accumulate).
// LDS staging: global_load_async_to_lds_b128 (ASYNCcnt) where available.
// ---------------------------------------------------------------------------

typedef __attribute__((ext_vector_type(16))) __bf16 v16bf;
typedef __attribute__((ext_vector_type(8)))  float  v8f;
typedef __attribute__((ext_vector_type(4)))  int    i32x4;

#define AS1 __attribute__((address_space(1)))
#define AS3 __attribute__((address_space(3)))

#if defined(__AMDGCN__) && __has_builtin(__builtin_amdgcn_global_load_async_to_lds_b128)
#define HAVE_ASYNC_LDS 1
#else
#define HAVE_ASYNC_LDS 0
#endif

union FragU {
  v16bf v;
  uint4 q[2];
};

__device__ inline unsigned short f2bf_bits(float f) {
  union { float f; unsigned u; } a; a.f = f;
  unsigned r = a.u + 0x7FFFu + ((a.u >> 16) & 1u);  // round-to-nearest-even
  return (unsigned short)(r >> 16);
}
__device__ inline float bf2f(unsigned short s) {
  union { unsigned u; float f; } a; a.u = ((unsigned)s) << 16;
  return a.f;
}

// 16-byte global -> LDS copy. Async (ASYNCcnt) on CDNA5 when the builtin is
// available; plain vector copy otherwise. Flat LDS addr[31:0] == LDS offset
// (ISA 10.2 aperture rules), so the integer cast to AS3 is value-correct.
__device__ inline void copy16_g2l(const unsigned short* g, unsigned short* l) {
#if HAVE_ASYNC_LDS
  __builtin_amdgcn_global_load_async_to_lds_b128(
      (AS1 i32x4*)(unsigned long long)g, (AS3 i32x4*)(unsigned long long)l, 0, 0);
#else
  *(uint4*)l = *(const uint4*)g;
#endif
}
__device__ inline void copy_join() {
#if HAVE_ASYNC_LDS
#if __has_builtin(__builtin_amdgcn_s_wait_asynccnt)
  __builtin_amdgcn_s_wait_asynccnt(0);
#else
  asm volatile("s_wait_asynccnt 0x0" ::: "memory");
#endif
#endif
}

// Load one WMMA operand fragment (16 bf16 per lane) from a row of 32 bf16.
// Lane-half 0 needs k = {0..7, 16..23}; half 1 needs k = {8..15, 24..31}:
// two contiguous 16B chunks each (CDNA5 16-bit operand layout).
__device__ inline v16bf load_frag_row32(const unsigned short* row, int half) {
  FragU f;
  f.q[0] = *(const uint4*)(row + half * 8);
  f.q[1] = *(const uint4*)(row + 16 + half * 8);
  return f.v;
}

// ---------------------------------------------------------------------------
// Pack f32 -> bf16 (zero-padding the tail so GEMM tiles never need guards).
// ---------------------------------------------------------------------------
__global__ __launch_bounds__(256)
void pack_bf16_kernel(const float* __restrict__ src, unsigned short* __restrict__ dst,
                      int valid, int total) {
  int i = blockIdx.x * 256 + threadIdx.x;
  if (i < total) dst[i] = f2bf_bits(i < valid ? src[i] : 0.0f);
}

#define BM 128
#define BN 64
#define BK 32

// ---------------------------------------------------------------------------
// GEMM:  Out[b][n][m] = act( Wbf[m][:] . X[:, n] + bias[m] )   (token-major out)
// mode 0: X = on-the-fly im2col of x[b][n][c] (3x3 conv, K = 192*9)
// mode 1: X = bf16 [1024][Ktot] token-major buffer (1x1 conv) -> async staging
// act  0: relu, 1: sigmoid.  Wbf rows padded so Mbase..Mbase+127 always valid.
// ---------------------------------------------------------------------------
__global__ __launch_bounds__(256)
void gemm_act_kernel(const unsigned short* __restrict__ Wbf,
                     const float* __restrict__ bias,
                     const float* __restrict__ Xconv,
                     const unsigned short* __restrict__ Xdir,
                     unsigned short* __restrict__ Out,
                     int M, int Ktot, int mode, int act) {
  __shared__ __align__(16) unsigned short Asm[BM * BK];  // [m][k]
  __shared__ __align__(16) unsigned short Bsm[BN * BK];  // [n][k]

  const int b     = blockIdx.z;
  const int Mbase = blockIdx.y * BM;
  const int Nbase = blockIdx.x * BN;
  const int tid   = threadIdx.x;
  const int lane  = tid & 31;
  const int wave  = tid >> 5;
  const int half  = lane >> 4;
  const int l16   = lane & 15;

  const float* xb = Xconv + (size_t)b * 1024 * 192;
  const unsigned short* xd = Xdir + (size_t)b * 1024 * (size_t)Ktot;
  unsigned short* outb = Out + (size_t)b * 1024 * (size_t)M;

  v8f acc[4];
#pragma unroll
  for (int i = 0; i < 4; ++i)
#pragma unroll
    for (int r = 0; r < 8; ++r) acc[i][r] = 0.0f;

  const int ksteps = Ktot / BK;
  for (int ks = 0; ks < ksteps; ++ks) {
    const int kb = ks * BK;
    __syncthreads();
    // --- stage A (bf16 weights, straight async copy: 128 rows x 64B) ---
#pragma unroll
    for (int rep = 0; rep < 2; ++rep) {
      int idx = rep * 256 + tid;            // 512 x 16B chunks
      int m = idx >> 2, part = idx & 3;
      copy16_g2l(Wbf + (size_t)(Mbase + m) * Ktot + kb + part * 8,
                 Asm + m * BK + part * 8);
    }
    // --- stage B ---
    if (mode == 0) {
      // im2col gather: k -> (c, dh, dw); n -> (row, col) in the 32x32 window
      for (int i = tid; i < BN * BK; i += 256) {
        int n = i & 63, k = i >> 6;
        int kg = kb + k;
        int c = kg / 9, rem = kg - c * 9;
        int dh = rem / 3 - 1, dw = rem - (rem / 3) * 3 - 1;
        int ng = Nbase + n;
        int rr = (ng >> 5) + dh, cc = (ng & 31) + dw;
        float v = 0.0f;
        if (rr >= 0 && rr < 32 && cc >= 0 && cc < 32)
          v = xb[(size_t)(rr * 32 + cc) * 192 + c];
        Bsm[n * BK + k] = f2bf_bits(v);
      }
    } else {
      // token-major X: 64 rows x 64B, straight async copy
      int n = tid >> 2, part = tid & 3;
      copy16_g2l(xd + (size_t)(Nbase + n) * Ktot + kb + part * 8,
                 Bsm + n * BK + part * 8);
    }
    if (ks + 1 < ksteps)  // global_prefetch_b8 hint for next weight tile
      __builtin_prefetch(&Wbf[(size_t)(Mbase + wave * 16) * Ktot + kb + BK], 0, 1);
    copy_join();
    __syncthreads();

    v16bf afrag = load_frag_row32(Asm + (wave * 16 + l16) * BK, half);
#pragma unroll
    for (int nb = 0; nb < 4; ++nb) {
      v16bf bfrag = load_frag_row32(Bsm + (nb * 16 + l16) * BK, half);
      acc[nb] = __builtin_amdgcn_wmma_f32_16x16x32_bf16(
          false, afrag, false, bfrag, (short)0, acc[nb], false, false);
    }
  }

#pragma unroll
  for (int nb = 0; nb < 4; ++nb) {
    int n = Nbase + nb * 16 + l16;
#pragma unroll
    for (int r = 0; r < 8; ++r) {
      int m = Mbase + wave * 16 + r + 8 * half;
      if (m < M) {
        float v = acc[nb][r] + bias[m];
        v = (act == 0) ? fmaxf(v, 0.0f) : 1.0f / (1.0f + __expf(-v));
        outb[(size_t)n * M + m] = f2bf_bits(v);
      }
    }
  }
}

// ---------------------------------------------------------------------------
// Build q/k/v [B][12][1024][16] bf16, replicating the reference's
// flatten-order index arithmetic. sig is token-major [1024][576].
// ---------------------------------------------------------------------------
__global__ __launch_bounds__(256)
void qkv_kernel(const float* __restrict__ x, const unsigned short* __restrict__ sig,
                unsigned short* __restrict__ q, unsigned short* __restrict__ k,
                unsigned short* __restrict__ v) {
  int idx = blockIdx.x * 256 + threadIdx.x;
  if (idx >= 16 * 12 * 1024 * 16) return;
  int d  = idx & 15;
  int n2 = (idx >> 4) & 1023;
  int bh = idx >> 14;
  int h  = bh % 12;
  int b  = bh / 12;

  int g2  = n2 * 192 + h * 16 + d;   // position in t_flat
  int ch2 = g2 >> 10;
  int n3  = g2 & 1023;
  int f   = n3 * 192 + ch2;          // position in prod flat [0, 196608)

  float yv = x[((size_t)b * 1024 + (f & 1023)) * 192 + (f >> 10)];
  const unsigned short* sb = sig + (size_t)b * 1024 * 576;
  int gq = f, gk = f + 196608, gv = f + 393216;
  float sq = bf2f(sb[(size_t)(gq / 576) * 576 + gq % 576]);
  float sk = bf2f(sb[(size_t)(gk / 576) * 576 + gk % 576]);
  float sv = bf2f(sb[(size_t)(gv / 576) * 576 + gv % 576]);
  q[idx] = f2bf_bits(sq * yv * 0.25f);  // SCALE = 16^-0.5
  k[idx] = f2bf_bits(sk * yv);
  v[idx] = f2bf_bits(sv * yv);
}

// ---------------------------------------------------------------------------
// Flash-style attention per (b, h): 128 query rows per block (16 per wave),
// streaming 64-key tiles. QK^T and P@V via WMMA (hd=16 zero-padded to K=32).
// Output O [B][1024][192] bf16 (token, h*16+d).
// ---------------------------------------------------------------------------
__global__ __launch_bounds__(256)
void attn_kernel(const unsigned short* __restrict__ Q,
                 const unsigned short* __restrict__ K,
                 const unsigned short* __restrict__ V,
                 unsigned short* __restrict__ O) {
  __shared__ __align__(16) unsigned short Ksm[64 * 32];     // [key][dpad]
  __shared__ __align__(16) unsigned short Vsm[16 * 64];     // [d][key]
  __shared__ __align__(16) unsigned short Psm[8][16 * 64];  // per-wave P [m][key]

  const int b = blockIdx.z, h = blockIdx.y, qt = blockIdx.x;
  const int tid = threadIdx.x, lane = tid & 31, wave = tid >> 5;
  const int half = lane >> 4, l16 = lane & 15;

  const size_t bh = ((size_t)b * 12 + h) * 1024 * 16;
  const unsigned short* Qb = Q + bh;
  const unsigned short* Kb = K + bh;
  const unsigned short* Vb = V + bh;

  // zero the d = 16..31 pad of Ksm once; in-loop staging only writes d < 16
  for (int i = tid; i < 64 * 16; i += 256) {
    int key = i >> 4, dd = i & 15;
    Ksm[key * 32 + 16 + dd] = 0;
  }

  const int qrow = qt * 128 + wave * 16 + l16;
  FragU qf;
  qf.q[0] = *(const uint4*)(Qb + (size_t)qrow * 16 + half * 8);
  qf.q[1] = make_uint4(0u, 0u, 0u, 0u);   // d 16..31 pad

  v8f oacc;
  float mi[8], li[8];
#pragma unroll
  for (int r = 0; r < 8; ++r) { oacc[r] = 0.0f; mi[r] = -1e30f; li[r] = 0.0f; }

  for (int kt = 0; kt < 16; ++kt) {
    const int kbase = kt * 64;
    __syncthreads();
    if (tid < 128) {
      int key = tid >> 1, part = tid & 1;
      // K tile: straight async copy of the 16 real d's per key row
      copy16_g2l(Kb + (size_t)(kbase + key) * 16 + part * 8,
                 Ksm + key * 32 + part * 8);
      // V tile: vector read, transposed scatter into [d][key]
      union { uint4 q; unsigned short s[8]; } u;
      u.q = *(const uint4*)(Vb + (size_t)(kbase + key) * 16 + part * 8);
#pragma unroll
      for (int j = 0; j < 8; ++j) Vsm[(part * 8 + j) * 64 + key] = u.s[j];
    }
    copy_join();
    __syncthreads();

    v8f s[4];
#pragma unroll
    for (int nb = 0; nb < 4; ++nb) {
#pragma unroll
      for (int r = 0; r < 8; ++r) s[nb][r] = 0.0f;
      v16bf kf = load_frag_row32(Ksm + (nb * 16 + l16) * 32, half);
      s[nb] = __builtin_amdgcn_wmma_f32_16x16x32_bf16(
          false, qf.v, false, kf, (short)0, s[nb], false, false);
    }

    // online softmax; row m = r + 8*half lives across 16 lanes x 4 accs
    float p[4][8];
#pragma unroll
    for (int r = 0; r < 8; ++r) {
      float rm = fmaxf(fmaxf(s[0][r], s[1][r]), fmaxf(s[2][r], s[3][r]));
      rm = fmaxf(rm, __shfl_xor(rm, 1));
      rm = fmaxf(rm, __shfl_xor(rm, 2));
      rm = fmaxf(rm, __shfl_xor(rm, 4));
      rm = fmaxf(rm, __shfl_xor(rm, 8));
      float newm = fmaxf(mi[r], rm);
      float rs = 0.0f;
#pragma unroll
      for (int nb = 0; nb < 4; ++nb) { p[nb][r] = __expf(s[nb][r] - newm); rs += p[nb][r]; }
      rs += __shfl_xor(rs, 1);
      rs += __shfl_xor(rs, 2);
      rs += __shfl_xor(rs, 4);
      rs += __shfl_xor(rs, 8);
      float sc = __expf(mi[r] - newm);
      li[r] = li[r] * sc + rs;
      oacc[r] *= sc;
      mi[r] = newm;
    }

    unsigned short* pw = Psm[wave];
#pragma unroll
    for (int nb = 0; nb < 4; ++nb)
#pragma unroll
      for (int r = 0; r < 8; ++r)
        pw[(r + 8 * half) * 64 + nb * 16 + l16] = f2bf_bits(p[nb][r]);
    __syncthreads();

#pragma unroll
    for (int c = 0; c < 2; ++c) {  // O += P(16x32) @ V(32x16)
      v16bf pf = load_frag_row32(pw + l16 * 64 + c * 32, half);
      v16bf vf = load_frag_row32(Vsm + l16 * 64 + c * 32, half);
      oacc = __builtin_amdgcn_wmma_f32_16x16x32_bf16(
          false, pf, false, vf, (short)0, oacc, false, false);
    }
  }

#pragma unroll
  for (int r = 0; r < 8; ++r) {
    int m = qt * 128 + wave * 16 + r + 8 * half;
    float ov = oacc[r] / li[r];
    O[((size_t)b * 1024 + m) * 192 + h * 16 + l16] = f2bf_bits(ov);
  }
}

// ---------------------------------------------------------------------------
// Final projection: out[t][c'] = sum_c O[t][c] * w_out[c'][c], fp32 output.
// ---------------------------------------------------------------------------
__global__ __launch_bounds__(256)
void proj_kernel(const unsigned short* __restrict__ Oin,   // [16384][192] bf16
                 const unsigned short* __restrict__ Wobf,  // [192][192] bf16
                 float* __restrict__ out) {
  __shared__ __align__(16) unsigned short Bsm[192 * 32];   // [c'][k-chunk]
  const int tid = threadIdx.x, lane = tid & 31, wave = tid >> 5;
  const int half = lane >> 4, l16 = lane & 15;
  const int Tbase = blockIdx.x * 128;

  v8f acc[12];
#pragma unroll
  for (int i = 0; i < 12; ++i)
#pragma unroll
    for (int r = 0; r < 8; ++r) acc[i][r] = 0.0f;

  for (int kc = 0; kc < 6; ++kc) {
    __syncthreads();
#pragma unroll
    for (int rep = 0; rep < 3; ++rep) {
      int idx = rep * 256 + tid;           // 768 x 16B chunks: 192 rows x 64B
      int n = idx >> 2, part = idx & 3;
      copy16_g2l(Wobf + (size_t)n * 192 + kc * 32 + part * 8,
                 Bsm + n * 32 + part * 8);
    }
    copy_join();
    __syncthreads();
    v16bf af = load_frag_row32(
        Oin + (size_t)(Tbase + wave * 16 + l16) * 192 + kc * 32, half);
#pragma unroll
    for (int nb = 0; nb < 12; ++nb) {
      v16bf bf = load_frag_row32(Bsm + (nb * 16 + l16) * 32, half);
      acc[nb] = __builtin_amdgcn_wmma_f32_16x16x32_bf16(
          false, af, false, bf, (short)0, acc[nb], false, false);
    }
  }
#pragma unroll
  for (int nb = 0; nb < 12; ++nb)
#pragma unroll
    for (int r = 0; r < 8; ++r)
      out[(size_t)(Tbase + wave * 16 + r + 8 * half) * 192 + nb * 16 + l16] =
          acc[nb][r];
}

// ---------------------------------------------------------------------------
extern "C" void kernel_launch(void* const* d_in, const int* in_sizes, int n_in,
                              void* d_out, int out_size, void* d_ws, size_t ws_size,
                              hipStream_t stream) {
  const float* x     = (const float*)d_in[0];  // [16][1024][192]
  const float* w1    = (const float*)d_in[1];  // [768][1728]
  const float* b1    = (const float*)d_in[2];
  const float* w2    = (const float*)d_in[3];  // [768][768]
  const float* b2    = (const float*)d_in[4];
  const float* w3    = (const float*)d_in[5];  // [576][768]
  const float* b3    = (const float*)d_in[6];
  const float* w_out = (const float*)d_in[7];  // [192][192]
  float* out = (float*)d_out;

  unsigned short* w1bf = (unsigned short*)d_ws;               // [768][1728]
  unsigned short* w2bf = w1bf + (size_t)768 * 1728;           // [768][768]
  unsigned short* w3bf = w2bf + (size_t)768 * 768;            // [640][768] (pad)
  unsigned short* wobf = w3bf + (size_t)640 * 768;            // [192][192]
  unsigned short* out1 = wobf + (size_t)192 * 192;            // [16][1024][768]
  unsigned short* out2 = out1 + (size_t)16 * 1024 * 768;      // [16][1024][768]
  unsigned short* out3 = out2 + (size_t)16 * 1024 * 768;      // [16][1024][576]
  unsigned short* qws  = out3 + (size_t)16 * 1024 * 576;      // [16][12][1024][16]
  unsigned short* kws  = qws  + (size_t)16 * 12 * 1024 * 16;
  unsigned short* vws  = kws  + (size_t)16 * 12 * 1024 * 16;
  unsigned short* ows  = vws  + (size_t)16 * 12 * 1024 * 16;  // [16][1024][192]

  dim3 blk(256);
  // weight pre-packing to bf16 (w3 zero-padded to 640 rows)
  pack_bf16_kernel<<<dim3((768 * 1728 + 255) / 256), blk, 0, stream>>>(
      w1, w1bf, 768 * 1728, 768 * 1728);
  pack_bf16_kernel<<<dim3((768 * 768 + 255) / 256), blk, 0, stream>>>(
      w2, w2bf, 768 * 768, 768 * 768);
  pack_bf16_kernel<<<dim3((640 * 768 + 255) / 256), blk, 0, stream>>>(
      w3, w3bf, 576 * 768, 640 * 768);
  pack_bf16_kernel<<<dim3((192 * 192 + 255) / 256), blk, 0, stream>>>(
      w_out, wobf, 192 * 192, 192 * 192);

  // conv3x3 (implicit GEMM) + relu
  gemm_act_kernel<<<dim3(16, 6, 16), blk, 0, stream>>>(
      w1bf, b1, x, out1 /*unused*/, out1, 768, 1728, 0, 0);
  // 1x1 768->768 + relu
  gemm_act_kernel<<<dim3(16, 6, 16), blk, 0, stream>>>(
      w2bf, b2, x /*unused*/, out1, out2, 768, 768, 1, 0);
  // 1x1 768->576 + sigmoid
  gemm_act_kernel<<<dim3(16, 5, 16), blk, 0, stream>>>(
      w3bf, b3, x /*unused*/, out2, out3, 576, 768, 1, 1);
  // elementwise q/k/v assembly
  qkv_kernel<<<dim3((16 * 12 * 1024 * 16) / 256), blk, 0, stream>>>(
      x, out3, qws, kws, vws);
  // attention
  attn_kernel<<<dim3(8, 12, 16), blk, 0, stream>>>(qws, kws, vws, ows);
  // output projection (fp32 result)
  proj_kernel<<<dim3(128), blk, 0, stream>>>(ows, wobf, out);
}
